// Sequence_36060545417781
// MI455X (gfx1250) — matrix operand order repaired and checked by reading, not compiled, not run on hardware
//
#include <hip/hip_runtime.h>
#include <hip/hip_bf16.h>
#include <math.h>

// Problem constants (from reference)
#define H     150
#define G     600          // 4*H gate rows
#define STEPS 10
#define KPAD  160          // K padded to 5 chunks of 32 for 16x16x32 WMMA
#define JPAD  608          // 38 tiles of 16 gate rows
#define NTILE 38
#define NWELEM (JPAD * KPAD)   // 97280 f16 elements per weight image

typedef _Float16 v16h  __attribute__((ext_vector_type(16)));
typedef _Float16 half8 __attribute__((ext_vector_type(8)));
typedef float    v8f   __attribute__((ext_vector_type(8)));
typedef float    f4    __attribute__((ext_vector_type(4)));

// d_ws layout (all offsets 16B-aligned):
//   [0,      38912)   XW    f32 [16][608]
//   [38912,  41344)   bsum  f32 [608]        (b_ih + b_hh, padded)
//   [41344, 235904)   Whh16 f16 [608][160]   (padded f16 image of W_hh)
//   [235904,430464)   Wih16 f16 [608][160]   (padded f16 image of W_ih)
//   [430464,435584)   Xs16  f16 [16][160]    (padded f16 image of xs slice)
#define WS_XW_OFF   0
#define WS_BS_OFF   38912
#define WS_WHH_OFF  41344
#define WS_WIH_OFF  235904
#define WS_XS_OFF   430464
#define PREP_TOTAL  (2 * NWELEM + 16 * KPAD + JPAD)   // 197728 elements

// ---------------------------------------------------------------------------
// Kernel 1: one-shot prep. Materializes zero-padded f16 images of W_hh, W_ih,
// the xs timestep slice, and the padded f32 bias sum. Fully coalesced,
// branch-light, so all later WMMA fragment feeds are unconditional b128 loads.
// ---------------------------------------------------------------------------
__global__ void __launch_bounds__(256)
k_prep(const float* __restrict__ inputx,
       const float* __restrict__ Wih,
       const float* __restrict__ Whh,
       const float* __restrict__ b_ih,
       const float* __restrict__ b_hh,
       const int*   __restrict__ xp,
       char* __restrict__ ws) {
    _Float16* Whh16 = (_Float16*)(ws + WS_WHH_OFF);
    _Float16* Wih16 = (_Float16*)(ws + WS_WIH_OFF);
    _Float16* Xs16  = (_Float16*)(ws + WS_XS_OFF);
    float*    bsum  = (float*)(ws + WS_BS_OFF);

    int idx = blockIdx.x * blockDim.x + threadIdx.x;
    if (idx < NWELEM) {
        int r = idx / KPAD, c = idx - r * KPAD;
        float v = (r < G && c < H) ? Whh[r * H + c] : 0.0f;
        Whh16[idx] = (_Float16)v;
    } else if (idx < 2 * NWELEM) {
        int i = idx - NWELEM;
        int r = i / KPAD, c = i - r * KPAD;
        float v = (r < G && c < H) ? Wih[r * H + c] : 0.0f;
        Wih16[i] = (_Float16)v;
    } else if (idx < 2 * NWELEM + 16 * KPAD) {
        int i = idx - 2 * NWELEM;
        int t = i / KPAD, k = i - t * KPAD;
        int xbase = xp[0] - STEPS;
        float v = (t < STEPS && k < H) ? inputx[(xbase + t) * H + k] : 0.0f;
        Xs16[i] = (_Float16)v;
    } else if (idx < PREP_TOTAL) {
        int j = idx - (2 * NWELEM + 16 * KPAD);
        bsum[j] = (j < G) ? (b_ih[j] + b_hh[j]) : 0.0f;
    }
}

// Build a v16h A-fragment (ISA 16-bit A layout) from a row-major f16 row:
//   a[0..7]  = row[K0 + half*8 + i],  a[8..15] = row[K0 + 16 + half*8 + i]
__device__ __forceinline__ v16h load_a_frag(const _Float16* row, int K0, int half) {
    half8 lo = *(const half8*)(row + K0 + half * 8);
    half8 hi = *(const half8*)(row + K0 + 16 + half * 8);
    v16h a;
#pragma unroll
    for (int i = 0; i < 8; ++i) { a[i] = lo[i]; a[8 + i] = hi[i]; }
    return a;
}

// Build a v16h B-fragment: b[0..7] = row[K0 + half*8 + i], b[8..15] = row[K0 + 16 + half*8 + i]
// (this lane's column, or replicated h when `row` is the shared h vector).
__device__ __forceinline__ v16h load_b_frag(const _Float16* row, int K0, int half) {
    half8 lo = *(const half8*)(row + K0 + half * 8);
    half8 hi = *(const half8*)(row + K0 + 16 + half * 8);
    v16h b;
#pragma unroll
    for (int i = 0; i < 8; ++i) { b[i] = lo[i]; b[8 + i] = hi[i]; }
    return b;
}

// ---------------------------------------------------------------------------
// Kernel 2: XW[t][j] = sum_k xs[t][k]*W_ih[j][k] + bsum[j]
// One wave32 per block, one 16-column gate tile per block. All fragment
// feeds are unconditional 16B vector loads from the padded images.
// ---------------------------------------------------------------------------
__global__ void __launch_bounds__(32)
k_xw(const char* __restrict__ ws_in, float* __restrict__ XW) {
    const _Float16* Wih16 = (const _Float16*)(ws_in + WS_WIH_OFF);
    const _Float16* Xs16  = (const _Float16*)(ws_in + WS_XS_OFF);
    const float*    bsum  = (const float*)(ws_in + WS_BS_OFF);

    const int lane = threadIdx.x;   // 0..31
    const int half = lane >> 4;
    const int nl   = lane & 15;
    const int j0   = blockIdx.x * 16;

    // C[m][n] = bsum[j0+n] for all m (n = lane&15)
    const float bs = bsum[j0 + nl];
    v8f acc;
#pragma unroll
    for (int v = 0; v < 8; ++v) acc[v] = bs;

    const _Float16* arow = Xs16 + nl * KPAD;           // A row: timestep t = nl
    const _Float16* brow = Wih16 + (j0 + nl) * KPAD;   // B col: gate j0 + nl

#pragma unroll
    for (int kc = 0; kc < 5; ++kc) {
        const int K0 = kc * 32;
        v16h a = load_a_frag(arow, K0, half);
        v16h b = load_b_frag(brow, K0, half);
        acc = __builtin_amdgcn_wmma_f32_16x16x32_f16(
                  false, a, false, b, (short)0, acc, false, false);
    }
    // D store: c[v] <-> (m = v + 8*half, n = lane&15); full padded tile
#pragma unroll
    for (int v = 0; v < 8; ++v) {
        XW[(v + 8 * half) * JPAD + j0 + nl] = acc[v];
    }
}

// ---------------------------------------------------------------------------
// Kernel 3: sequential recurrence, single workgroup (8 wave32 waves).
// XW panel is preloaded into LDS once; per-step traffic is only the
// L2-resident f16 W_hh image (A-frags) + LDS h vector (B-frags, broadcast).
// ---------------------------------------------------------------------------
__global__ void __launch_bounds__(256)
k_rec(const float* __restrict__ XW,
      const _Float16* __restrict__ W16,
      float* __restrict__ out) {
    __shared__ alignas(16) float    xwS[16 * JPAD];  // 38,912 B
    __shared__ float                gates[JPAD];     //  2,432 B
    __shared__ alignas(16) _Float16 h16[KPAD];       //    320 B

    const int tid  = threadIdx.x;
    const int lane = tid & 31;
    const int wave = tid >> 5;
    const int half = lane >> 4;
    const int nl   = lane & 15;

    for (int i = tid; i < 16 * JPAD; i += 256) xwS[i] = XW[i];
    if (tid < KPAD) h16[tid] = (_Float16)0.0f;       // h0 = 0 (incl. padding)
    float cpriv = 0.0f;                              // c0 = 0, private to thread j
    __syncthreads();

    for (int t = 0; t < STEPS; ++t) {
        const float* XWt = xwS + t * JPAD;
        for (int jt = wave; jt < NTILE; jt += 8) {
            const int j0 = jt * 16;
            // C init: c[v] = XW[t][j0 + v + 8*half] (uniform across columns)
            v8f acc;
            const f4* cp = (const f4*)(XWt + j0 + 8 * half);
            f4 c0 = cp[0], c1 = cp[1];
#pragma unroll
            for (int v = 0; v < 4; ++v) { acc[v] = c0[v]; acc[4 + v] = c1[v]; }

            const _Float16* wrow = W16 + (j0 + nl) * KPAD;  // A row (L2 hit)
#pragma unroll
            for (int kc = 0; kc < 5; ++kc) {
                const int K0 = kc * 32;
                v16h a = load_a_frag(wrow, K0, half);
                v16h b = load_b_frag(h16, K0, half);  // h replicated over N
                acc = __builtin_amdgcn_wmma_f32_16x16x32_f16(
                          false, a, false, b, (short)0, acc, false, false);
            }
            // Every column of D is identical; lanes n==0 (lanes 0 and 16)
            // hold M=0..7 / M=8..15 in acc[0..7] -> write gates to LDS.
            if (nl == 0) {
#pragma unroll
                for (int v = 0; v < 8; ++v) {
                    int j = j0 + 8 * half + v;
                    if (j < G) gates[j] = acc[v];
                }
            }
        }
        __syncthreads();

        if (tid < H) {
            float zi = gates[tid];
            float zf = gates[H + tid];
            float zg = gates[2 * H + tid];
            float zo = gates[3 * H + tid];
            float gi = 1.0f / (1.0f + expf(-zi));
            float gf = 1.0f / (1.0f + expf(-zf));
            float gg = tanhf(zg);
            float go = 1.0f / (1.0f + expf(-zo));
            cpriv = gf * cpriv + gi * gg;
            float hnew = go * tanhf(cpriv);
            h16[tid] = (_Float16)hnew;           // feed next step's matvec
            if (t == STEPS - 1) out[tid] = hnew; // final h -> d_out (150 f32)
        }
        __syncthreads();
    }
}

// ---------------------------------------------------------------------------
// Launcher: three stream-ordered kernels; graph-capture safe, deterministic.
// Inputs: inputx(2400 f32), W_ih(90000 f32), W_hh(90000 f32),
//         b_ih(600 f32), b_hh(600 f32), x(1 i32)
// ---------------------------------------------------------------------------
extern "C" void kernel_launch(void* const* d_in, const int* in_sizes, int n_in,
                              void* d_out, int out_size, void* d_ws, size_t ws_size,
                              hipStream_t stream) {
    const float* inputx = (const float*)d_in[0];
    const float* W_ih   = (const float*)d_in[1];
    const float* W_hh   = (const float*)d_in[2];
    const float* b_ih   = (const float*)d_in[3];
    const float* b_hh   = (const float*)d_in[4];
    const int*   xp     = (const int*)d_in[5];

    char*     ws  = (char*)d_ws;
    float*    XW  = (float*)(ws + WS_XW_OFF);
    _Float16* W16 = (_Float16*)(ws + WS_WHH_OFF);

    k_prep<<<(PREP_TOTAL + 255) / 256, 256, 0, stream>>>(
        inputx, W_ih, W_hh, b_ih, b_hh, xp, ws);
    k_xw<<<NTILE, 32, 0, stream>>>(ws, XW);
    k_rec<<<1, 256, 0, stream>>>(XW, W16, (float*)d_out);
}
